// Air3DCNFROM_51634096832880
// MI455X (gfx1250) — matrix operand order repaired and checked by reading, not compiled, not run on hardware
//
#include <hip/hip_runtime.h>
#include <hip/hip_bf16.h>
#include <math.h>

typedef _Float16 v16h __attribute__((ext_vector_type(16)));
typedef _Float16 v8h  __attribute__((ext_vector_type(8)));
typedef float    v8f  __attribute__((ext_vector_type(8)));

#define AS3 __attribute__((address_space(3)))
#define AS1 __attribute__((address_space(1)))

#define BATCH   65536
#define LATENT  10
#define STEPS   101
#define HID     512
#define MT      64          // batch rows per block
#define AST     520         // LDS activation row stride (halves), 1040B = 16B-aligned rows

#if __has_builtin(__builtin_amdgcn_tanhf)
__device__ __forceinline__ float fast_tanh(float v) { return __builtin_amdgcn_tanhf(v); }
#else
__device__ __forceinline__ float fast_tanh(float v) { return tanhf(v); }
#endif

// ---------------------------------------------------------------------------
// Kernel 1: shared latent trajectory. alpha0 == 0 and f(t,a) is batch-
// independent, so traj is [STEPS][LATENT] computed once by one workgroup.
// ---------------------------------------------------------------------------
__global__ __launch_bounds__(128)
void traj_kernel(const float* __restrict__ pn_w0, const float* __restrict__ pn_b0,
                 const float* __restrict__ pn_w1, const float* __restrict__ pn_b1,
                 const float* __restrict__ pn_w2, const float* __restrict__ pn_b2,
                 float* __restrict__ traj)
{
    __shared__ float w0s[11 * 128];
    __shared__ float w1s[128 * 128];
    __shared__ float w2s[128 * LATENT];
    __shared__ float b0s[128], b1s[128], b2s[16];
    __shared__ float h0[128], h1[128];
    __shared__ float a[16], ain[16], k1[16], k2[16], k3[16], k4[16];

    const int tid = threadIdx.x;
    for (int i = tid; i < 11 * 128; i += 128)  w0s[i] = pn_w0[i];
    for (int i = tid; i < 128 * 128; i += 128) w1s[i] = pn_w1[i];
    for (int i = tid; i < 128 * LATENT; i += 128) w2s[i] = pn_w2[i];
    b0s[tid] = pn_b0[tid];
    b1s[tid] = pn_b1[tid];
    if (tid < LATENT) { b2s[tid] = pn_b2[tid]; a[tid] = 0.0f; traj[tid] = 0.0f; }
    __syncthreads();

    const float dt = 1.0f / (float)(STEPS - 1);

    // evaluates dst = f(tin, ain); all 128 threads participate
    auto evalf = [&](float tin, float* dst) {
        float acc = b0s[tid];
        #pragma unroll
        for (int i = 0; i < LATENT; ++i) acc += ain[i] * w0s[i * 128 + tid];
        acc += tin * w0s[LATENT * 128 + tid];
        h0[tid] = fast_tanh(acc);
        __syncthreads();
        float acc1 = b1s[tid];
        for (int i = 0; i < 128; ++i) acc1 += h0[i] * w1s[i * 128 + tid];
        h1[tid] = fast_tanh(acc1);
        __syncthreads();
        if (tid < LATENT) {
            float o = b2s[tid];
            for (int i = 0; i < 128; ++i) o += h1[i] * w2s[i * LATENT + tid];
            dst[tid] = o;
        }
        __syncthreads();
    };

    for (int s = 0; s < STEPS - 1; ++s) {
        float t = dt * (float)s;
        if (tid < LATENT) ain[tid] = a[tid];
        __syncthreads();
        evalf(t, k1);
        if (tid < LATENT) ain[tid] = a[tid] + 0.5f * dt * k1[tid];
        __syncthreads();
        evalf(t + 0.5f * dt, k2);
        if (tid < LATENT) ain[tid] = a[tid] + 0.5f * dt * k2[tid];
        __syncthreads();
        evalf(t + 0.5f * dt, k3);
        if (tid < LATENT) ain[tid] = a[tid] + dt * k3[tid];
        __syncthreads();
        evalf(t + dt, k4);
        if (tid < LATENT) {
            float an = a[tid] + (dt / 6.0f) * (k1[tid] + 2.0f * k2[tid] + 2.0f * k3[tid] + k4[tid]);
            a[tid] = an;
            traj[(s + 1) * LATENT + tid] = an;
        }
        __syncthreads();
    }
}

// ---------------------------------------------------------------------------
// Kernel 2: repack a [Ksrc x 512] f32 weight into f16 WMMA B-fragments.
// Fragment f = kt*32 + nt covers K in [kt*32, kt*32+32), N in [nt*16, nt*16+16).
// Within a fragment: element lane*16 + j -> N = nt*16 + lane%16,
//                                          K = kt*32 + (lane/16)*16 + j.
// K rows >= Ksrc are zero-padded (dec_w0 padded 106 -> 128).
// ---------------------------------------------------------------------------
__global__ __launch_bounds__(256)
void pack_kernel(const float* __restrict__ src, _Float16* __restrict__ dst,
                 int Ksrc, int Kp)
{
    int total = Kp * HID;
    for (int e = blockIdx.x * blockDim.x + threadIdx.x; e < total;
         e += gridDim.x * blockDim.x) {
        int f      = e >> 9;        // /512 elements per fragment
        int within = e & 511;
        int lane   = within >> 4;
        int j      = within & 15;
        int kt     = f >> 5;
        int nt     = f & 31;
        int n = nt * 16 + (lane & 15);
        int k = kt * 32 + ((lane >> 4) << 4) + j;
        float v = (k < Ksrc) ? src[k * HID + n] : 0.0f;
        dst[e] = (_Float16)v;
    }
}

// ---------------------------------------------------------------------------
// One hidden layer: dst = tanh(src @ W + b). Explicit address spaces so the
// compiler emits ds_load_b128 (no generic null checks) and global_load_b128
// (no flat). KSTEPS is compile-time so every address is base + imm offset.
// 2(N)x4(M) register blocking: 8 wmma per (8 ds_load_b128 + 4 global b128).
// ---------------------------------------------------------------------------
template<int KSTEPS>
__device__ __forceinline__
void gemm_layer(const AS3 _Float16* src, AS3 _Float16* dst,
                const AS1 _Float16* wp, const float* bias,
                int wave, int lane)
{
    const int ntBase = wave * 4;                // each wave owns 4 of 32 N-tiles
    #pragma unroll
    for (int p = 0; p < 2; ++p) {               // two N-pairs
        const int nt0 = ntBase + 2 * p;
        v8f acc[2][4];
        #pragma unroll
        for (int n2 = 0; n2 < 2; ++n2)
            #pragma unroll
            for (int m4 = 0; m4 < 4; ++m4)
                acc[n2][m4] = (v8f){0.f, 0.f, 0.f, 0.f, 0.f, 0.f, 0.f, 0.f};

        const int arowL = lane & 15;
        const int koffL = (lane >> 4) << 4;
        const AS3 _Float16* abase = src + arowL * AST + koffL;
        const AS1 _Float16* bbase = wp + (size_t)nt0 * 512 + lane * 16;

        #pragma unroll
        for (int ks = 0; ks < KSTEPS; ++ks) {
            // A fragments (16x32 f16) for all 4 row tiles: lanes<16 hold
            // K {0..15}, lanes>=16 hold K {16..31}.
            v16h av[4];
            #pragma unroll
            for (int m4 = 0; m4 < 4; ++m4) {
                v8h a0 = *(const AS3 v8h*)(abase + m4 * 16 * AST + ks * 32);
                v8h a1 = *(const AS3 v8h*)(abase + m4 * 16 * AST + ks * 32 + 8);
                av[m4] = __builtin_shufflevector(a0, a1,
                          0, 1, 2, 3, 4, 5, 6, 7, 8, 9, 10, 11, 12, 13, 14, 15);
            }
            // B fragments for the 2 column tiles (pre-packed, 32B/lane).
            v16h bv[2];
            #pragma unroll
            for (int n2 = 0; n2 < 2; ++n2) {
                const AS1 _Float16* bp = bbase + (size_t)(ks * 32 + n2) * 512;
                v8h b0 = *(const AS1 v8h*)bp;
                v8h b1 = *(const AS1 v8h*)(bp + 8);
                bv[n2] = __builtin_shufflevector(b0, b1,
                          0, 1, 2, 3, 4, 5, 6, 7, 8, 9, 10, 11, 12, 13, 14, 15);
            }
            #pragma unroll
            for (int n2 = 0; n2 < 2; ++n2)
                #pragma unroll
                for (int m4 = 0; m4 < 4; ++m4)
                    acc[n2][m4] = __builtin_amdgcn_wmma_f32_16x16x32_f16(
                        false, av[m4], false, bv[n2], (short)0, acc[n2][m4],
                        false, false);
        }
        // epilogue: bias + tanh, store f16 to LDS for next layer.
        #pragma unroll
        for (int n2 = 0; n2 < 2; ++n2) {
            const int n   = (nt0 + n2) * 16 + (lane & 15);
            const float bvs = bias[n];
            #pragma unroll
            for (int m4 = 0; m4 < 4; ++m4) {
                const int mBase = m4 * 16 + ((lane >> 4) << 3);
                #pragma unroll
                for (int r8 = 0; r8 < 8; ++r8) {
                    float v = fast_tanh(acc[n2][m4][r8] + bvs);
                    dst[(mBase + r8) * AST + n] = (_Float16)v;
                }
            }
        }
    }
}

// ---------------------------------------------------------------------------
// Kernel 3: main decoder. 64-row tile per block, 8 waves, f16 WMMA with f32
// accumulation, activations ping-pong in LDS, weights streamed from L2.
// ---------------------------------------------------------------------------
__global__ __launch_bounds__(256)
void decoder_kernel(const float* __restrict__ x, const float* __restrict__ tau,
                    const float* __restrict__ traj,
                    const _Float16* __restrict__ w0p, const float* __restrict__ b0,
                    const _Float16* __restrict__ w1p, const float* __restrict__ b1,
                    const _Float16* __restrict__ w2p, const float* __restrict__ b2,
                    const float* __restrict__ w3, const float* __restrict__ b3,
                    float* __restrict__ out)
{
    __shared__ _Float16 bufA[MT * AST];
    __shared__ _Float16 bufB[MT * AST];
    __shared__ float    red[MT * 4];

    const int tid     = threadIdx.x;
    const int lane    = tid & 31;
    const int wave    = tid >> 5;               // 0..7
    const int rowBase = blockIdx.x * MT;

    AS3 _Float16* ldsA = (AS3 _Float16*)bufA;
    AS3 _Float16* ldsB = (AS3 _Float16*)bufB;

    // ---- stage layer-0 input: [Fourier(96) | alpha-lerp(10) | zero-pad(22)] ----
    for (int e = tid; e < MT * 128; e += 256) {
        int r  = e >> 7;
        int c  = e & 127;
        int gr = rowBase + r;
        float v = 0.0f;
        if (c < 96) {
            int d  = c >> 5;                    // which coordinate 0..2
            int cc = c & 31;                    // within coord: 16 sin then 16 cos
            int j  = cc & 15;
            float freq = 1.0f + 0.6f * (float)j;            // linspace(1,10,16)
            float p = 6.283185307179586f * x[gr * 3 + d] * freq;
            v = (cc < 16) ? __sinf(p) : __cosf(p);
        } else if (c < 106) {
            int m = c - 96;
            float ft = tau[gr] * (float)(STEPS - 1);        // tau / dtau
            int idx = (int)floorf(ft);
            idx = idx < 0 ? 0 : (idx > STEPS - 2 ? STEPS - 2 : idx);
            float ratio = ft - (float)idx;
            float a0 = traj[idx * LATENT + m];
            float a1 = traj[(idx + 1) * LATENT + m];
            v = a0 + ratio * (a1 - a0);
        }
        ldsA[r * AST + c] = (_Float16)v;
    }
    __syncthreads();

    gemm_layer<4>(ldsA, ldsB, (const AS1 _Float16*)w0p, b0, wave, lane);   // 128->512
    __syncthreads();
    gemm_layer<16>(ldsB, ldsA, (const AS1 _Float16*)w1p, b1, wave, lane);  // 512->512
    __syncthreads();
    gemm_layer<16>(ldsA, ldsB, (const AS1 _Float16*)w2p, b2, wave, lane);  // 512->512
    __syncthreads();

    // ---- final 512 -> 1 + boundary-condition epilogue ----
    for (int q = tid; q < MT * 4; q += 256) {
        int r = q >> 2;
        int p = q & 3;
        float s = 0.0f;
        int base = p * 128;
        #pragma unroll
        for (int v8 = 0; v8 < 16; ++v8) {
            v8h hv = *(const AS3 v8h*)(ldsB + r * AST + base + v8 * 8);
            #pragma unroll
            for (int i = 0; i < 8; ++i)
                s += (float)hv[i] * w3[base + v8 * 8 + i];
        }
        red[q] = s;
    }
    __syncthreads();
    for (int r = tid; r < MT; r += 256) {
        float u = red[r * 4] + red[r * 4 + 1] + red[r * 4 + 2] + red[r * 4 + 3] + b3[0];
        int gr = rowBase + r;
        float x0 = x[gr * 3 + 0];
        float x1 = x[gr * 3 + 1];
        float lx = sqrtf(x0 * x0 + x1 * x1) - 0.25f;
        out[gr] = lx + tau[gr] * u;
    }
}

// ---------------------------------------------------------------------------
extern "C" void kernel_launch(void* const* d_in, const int* in_sizes, int n_in,
                              void* d_out, int out_size, void* d_ws, size_t ws_size,
                              hipStream_t stream) {
    const float* x      = (const float*)d_in[0];
    const float* tau    = (const float*)d_in[1];
    const float* dec_w0 = (const float*)d_in[2];
    const float* dec_b0 = (const float*)d_in[3];
    const float* dec_w1 = (const float*)d_in[4];
    const float* dec_b1 = (const float*)d_in[5];
    const float* dec_w2 = (const float*)d_in[6];
    const float* dec_b2 = (const float*)d_in[7];
    const float* dec_w3 = (const float*)d_in[8];
    const float* dec_b3 = (const float*)d_in[9];
    const float* pn_w0  = (const float*)d_in[10];
    const float* pn_b0  = (const float*)d_in[11];
    const float* pn_w1  = (const float*)d_in[12];
    const float* pn_b1  = (const float*)d_in[13];
    const float* pn_w2  = (const float*)d_in[14];
    const float* pn_b2  = (const float*)d_in[15];
    float* out = (float*)d_out;

    // workspace layout
    char* ws = (char*)d_ws;
    float*    trajWs = (float*)ws;                              //   4 KB
    _Float16* w0p = (_Float16*)(ws + 4096);                     // 128 KB (128x512)
    _Float16* w1p = (_Float16*)(ws + 4096 + 131072);            // 512 KB (512x512)
    _Float16* w2p = (_Float16*)(ws + 4096 + 131072 + 524288);   // 512 KB (512x512)

    traj_kernel<<<1, 128, 0, stream>>>(pn_w0, pn_b0, pn_w1, pn_b1, pn_w2, pn_b2, trajWs);
    pack_kernel<<<128, 256, 0, stream>>>(dec_w0, w0p, 106, 128);
    pack_kernel<<<512, 256, 0, stream>>>(dec_w1, w1p, 512, 512);
    pack_kernel<<<512, 256, 0, stream>>>(dec_w2, w2p, 512, 512);
    decoder_kernel<<<BATCH / MT, 256, 0, stream>>>(
        x, tau, trajWs,
        w0p, dec_b0, w1p, dec_b1, w2p, dec_b2,
        dec_w3, dec_b3, out);
}
